// LSTM_89567247991391
// MI455X (gfx1250) — matrix operand order repaired and checked by reading, not compiled, not run on hardware
//
#include <hip/hip_runtime.h>

// ---------------------------------------------------------------------------
// Types for CDNA5 WMMA (wave32): A/B = v16h (16 x f16 = 8 VGPRs), C/D = v8f
// ---------------------------------------------------------------------------
typedef __attribute__((ext_vector_type(16))) _Float16 v16h;
typedef __attribute__((ext_vector_type(8)))  float    v8f;
typedef __attribute__((ext_vector_type(4)))  unsigned u32x4;
typedef __attribute__((ext_vector_type(4)))  float    f32x4;
typedef __attribute__((ext_vector_type(4)))  _Float16 h16x4;

union Frag { v16h v; u32x4 u[2]; };

#define T_STEPS 512
#define BATCH   64
#define IN_DIM  256
#define HID     512
#define OUT_DIM 256
#define G4H     2048          // 4*HID gate columns
#define NWG_REC 32            // persistent workgroups for the scan

static __device__ __forceinline__ v8f wmma_f16(v16h a, v16h b, v8f c) {
  // D = A(16x32 f16) * B(32x16 f16) + C(16x16 f32)
  return __builtin_amdgcn_wmma_f32_16x16x32_f16(false, a, false, b, (short)0, c,
                                                false, false);
}

static __device__ __forceinline__ float sigm_(float x) {
  return 1.0f / (1.0f + __expf(-x));
}
static __device__ __forceinline__ float tanh_(float x) {
  float e = __expf(2.0f * x);           // saturates correctly at +/-inf
  return 1.0f - 2.0f / (e + 1.0f);
}

// ---------------------------------------------------------------------------
// Software grid barrier (NWG_REC co-resident WGs; 144KB LDS/WG => <=2 per WGP)
// bar[0] = arrive counter, bar[1] = generation
// ---------------------------------------------------------------------------
static __device__ __forceinline__ void grid_barrier(unsigned* bar) {
  __syncthreads();
  if (threadIdx.x == 0) {
    __threadfence();                                    // release my stores
    unsigned gen = __atomic_load_n(bar + 1, __ATOMIC_RELAXED);
    if (atomicAdd(bar, 1u) == NWG_REC - 1u) {
      __atomic_store_n(bar, 0u, __ATOMIC_RELAXED);
      __threadfence();
      __atomic_store_n(bar + 1, gen + 1u, __ATOMIC_RELEASE);
    } else {
      while (__atomic_load_n(bar + 1, __ATOMIC_ACQUIRE) == gen)
        __builtin_amdgcn_s_sleep(1);
    }
    __threadfence();                                    // acquire others' stores
  }
  __syncthreads();
}

// ---------------------------------------------------------------------------
// Init: zero barrier state and h_0 (must run every launch; ws is poisoned)
// ---------------------------------------------------------------------------
__global__ __launch_bounds__(256) void init_state(unsigned* bar, _Float16* h0) {
  int i = blockIdx.x * 256 + threadIdx.x;
  if (i < 2) bar[i] = 0u;
  for (int j = i; j < BATCH * HID; j += gridDim.x * 256) h0[j] = (_Float16)0.0f;
}

// ---------------------------------------------------------------------------
// fp32 -> f16 elementwise (vectorized x4)
// ---------------------------------------------------------------------------
__global__ __launch_bounds__(256) void cvt_f32_to_f16(const float* __restrict__ src,
                                                      _Float16* __restrict__ dst,
                                                      int n4) {
  int i = blockIdx.x * 256 + threadIdx.x;
  if (i >= n4) return;
  f32x4 v = ((const f32x4*)src)[i];
  h16x4 h;
  h[0] = (_Float16)v[0]; h[1] = (_Float16)v[1];
  h[2] = (_Float16)v[2]; h[3] = (_Float16)v[3];
  ((h16x4*)dst)[i] = h;
}

// ---------------------------------------------------------------------------
// Pack W[N][K] fp32 row-major into this file's B-fragment order (contiguous k per lane half, same as A loads):
// frag(nt,kt) = 32 lanes x 16 halves; lane<16: N=nt*16+lane, K=kt*32+h
//                                     lane>=16: N=nt*16+lane-16, K=kt*32+16+h
// ---------------------------------------------------------------------------
__global__ __launch_bounds__(256) void pack_wB(const float* __restrict__ W,
                                               _Float16* __restrict__ out,
                                               int N, int K) {
  int p = blockIdx.x * 256 + threadIdx.x;
  if (p >= N * K) return;
  int KT   = K >> 5;
  int h    = p & 15;
  int lane = (p >> 4) & 31;
  int fi   = p >> 9;
  int kt   = fi % KT;
  int nt   = fi / KT;
  int n    = (nt << 4) + (lane & 15);
  int k    = (kt << 5) + ((lane >> 4) << 4) + h;
  out[p] = (_Float16)W[(size_t)n * K + k];
}

// ---------------------------------------------------------------------------
// Generic WMMA GEMM: C[M,N] = A[M,K](f16) * Bpacked + biasA + biasB
// Block 128 thr = 4 waves (2Mx2N), wave tile 32x64 (2x4 WMMA tiles).
// Everything streamed from L2 (X, weights, hs all fit in 192MB L2).
// ---------------------------------------------------------------------------
__global__ __launch_bounds__(128) void wmma_gemm(
    const _Float16* __restrict__ A, const _Float16* __restrict__ Bp,
    float* __restrict__ Cf, _Float16* __restrict__ Ch,
    const float* __restrict__ biasA, const float* __restrict__ biasB,
    int M, int N, int K) {
  const int lane  = threadIdx.x & 31;
  const int wid   = threadIdx.x >> 5;
  const int waveN = wid & 1, waveM = wid >> 1;
  const int m0    = blockIdx.x * 64 + waveM * 32;
  const int n0    = blockIdx.y * 128 + waveN * 64;
  const int KT    = K >> 5;
  const int laneM = lane & 15;
  const int hiK   = (lane >> 4) << 4;   // A K-phase: 0 or 16

  const v8f vzero = {};
  v8f acc[2][4];
#pragma unroll
  for (int i = 0; i < 2; ++i)
#pragma unroll
    for (int j = 0; j < 4; ++j) acc[i][j] = vzero;

  const _Float16* a0p   = A + (size_t)(m0 + laneM) * K;
  const _Float16* a1p   = a0p + (size_t)16 * K;
  const _Float16* bbase = Bp + (((size_t)(n0 >> 4) * KT) << 9) + (lane << 4);

  for (int kt = 0; kt < KT; ++kt) {
    Frag a0, a1;
    const int kb = (kt << 5) + hiK;
    a0.u[0] = *(const u32x4*)(a0p + kb);
    a0.u[1] = *(const u32x4*)(a0p + kb + 8);
    a1.u[0] = *(const u32x4*)(a1p + kb);
    a1.u[1] = *(const u32x4*)(a1p + kb + 8);
#pragma unroll
    for (int j = 0; j < 4; ++j) {
      Frag b;
      const _Float16* bp = bbase + (((size_t)(j * KT + kt)) << 9);
      b.u[0] = *(const u32x4*)bp;
      b.u[1] = *(const u32x4*)(bp + 8);
      acc[0][j] = wmma_f16(a0.v, b.v, acc[0][j]);
      acc[1][j] = wmma_f16(a1.v, b.v, acc[1][j]);
    }
  }

  const int rOff = (lane >> 4) << 3;    // C rows: lanes 0-15 -> M+0..7, 16-31 -> M+8..15
#pragma unroll
  for (int j = 0; j < 4; ++j) {
    const int n = n0 + j * 16 + laneM;
    float bv = 0.0f;
    if (biasA) bv += biasA[n];
    if (biasB) bv += biasB[n];
#pragma unroll
    for (int mt = 0; mt < 2; ++mt) {
      const int rbase = m0 + mt * 16 + rOff;
#pragma unroll
      for (int r = 0; r < 8; ++r) {
        float val = acc[mt][j][r] + bv;
        if (Ch) Ch[(size_t)(rbase + r) * N + n] = (_Float16)val;
        else    Cf[(size_t)(rbase + r) * N + n] = val;
      }
    }
  }
}

// ---------------------------------------------------------------------------
// Persistent recurrent scan. 32 WGs x 256 thr; WG s owns hidden units
// [s*16, s*16+16) => gate columns {g*512 + s*16 + j}. Its packed W_hh slice
// (64KB) lives in LDS for all 512 steps; per step it pulls h_t (64KB f16)
// from L2, runs 256 WMMAs, applies gates in fp32, writes its h_{t+1} slice.
// ---------------------------------------------------------------------------
__global__ __launch_bounds__(256) void lstm_recurrent(
    const _Float16* __restrict__ xg,     // [T][B][4H] f16, biases folded in
    const _Float16* __restrict__ whhP,   // fragment-packed [128 nt][16 kt][512]
    _Float16* __restrict__ hseq,         // [(T+1)][B][H] f16, hseq[0]=0
    float* __restrict__ hT, float* __restrict__ cT,
    unsigned* __restrict__ bar) {
  __shared__ _Float16 s_w[4 * 16 * 512];   // 64KB: packed W slice [gate][kt][512]
  __shared__ _Float16 s_h[BATCH * HID];    // 64KB: h_t broadcast
  __shared__ float    s_g[BATCH * 64];     // 16KB: raw gate GEMM result

  const int s    = blockIdx.x;
  const int tid  = threadIdx.x;
  const int lane = tid & 31;
  const int wid  = tid >> 5;

  // Stage this slice's packed W_hh: 4 gates x 16KB, already fragment-ordered.
#pragma unroll
  for (int g = 0; g < 4; ++g) {
    const u32x4* src = (const u32x4*)(whhP + (((size_t)(g * 32 + s) * 16) << 9));
    u32x4*       dst = (u32x4*)(s_w + (((size_t)g * 16) << 9));
    for (int i = tid; i < 1024; i += 256) dst[i] = src[i];
  }

  const int nt    = wid & 3;           // local N-tile (== gate index)
  const int mtb   = (wid >> 2) << 1;   // M-tile base: 0 or 2
  const int laneM = lane & 15;
  const int hiK   = (lane >> 4) << 4;
  const int rOff  = (lane >> 4) << 3;
  const v8f vzero = {};

  float c_reg[4] = {0.f, 0.f, 0.f, 0.f};   // cell state lives in VGPRs

  for (int t = 0; t < T_STEPS; ++t) {
    // Broadcast h_t into LDS (64KB, 256B/thread).
    const u32x4* hsrc = (const u32x4*)(hseq + (size_t)t * (BATCH * HID));
    u32x4*       hdst = (u32x4*)s_h;
    for (int i = tid; i < (BATCH * HID) / 8; i += 256) hdst[i] = hsrc[i];
    __builtin_prefetch(xg + (size_t)t * (BATCH * G4H) + (size_t)tid * 512, 0, 1);
    __syncthreads();

    // h_t[64x512] x W_hh_slice[512x64] -> s_g[64x64], two acc chains per wave.
    v8f acc0 = vzero, acc1 = vzero;
    const _Float16* ar0   = s_h + (size_t)(mtb * 16 + laneM) * HID;
    const _Float16* ar1   = ar0 + (size_t)16 * HID;
    const _Float16* bbase = s_w + (((size_t)nt * 16) << 9) + (lane << 4);
#pragma unroll 4
    for (int kt = 0; kt < 16; ++kt) {
      Frag b, a0, a1;
      const _Float16* bp = bbase + ((size_t)kt << 9);
      b.u[0] = *(const u32x4*)bp;
      b.u[1] = *(const u32x4*)(bp + 8);
      const int kb = (kt << 5) + hiK;
      a0.u[0] = *(const u32x4*)(ar0 + kb);
      a0.u[1] = *(const u32x4*)(ar0 + kb + 8);
      a1.u[0] = *(const u32x4*)(ar1 + kb);
      a1.u[1] = *(const u32x4*)(ar1 + kb + 8);
      acc0 = wmma_f16(a0.v, b.v, acc0);
      acc1 = wmma_f16(a1.v, b.v, acc1);
    }
    {
      const int n  = (nt << 4) + laneM;
      const int r0 = (mtb << 4) + rOff;
#pragma unroll
      for (int r = 0; r < 8; ++r) s_g[(r0 + r) * 64 + n] = acc0[r];
      const int r1 = r0 + 16;
#pragma unroll
      for (int r = 0; r < 8; ++r) s_g[(r1 + r) * 64 + n] = acc1[r];
    }
    __syncthreads();

    // Gates + state update: each thread owns 4 (batch, hidden) cells.
    const _Float16* xgr = xg + (size_t)t * (BATCH * G4H);
    _Float16*       hn  = hseq + (size_t)(t + 1) * (BATCH * HID);
#pragma unroll
    for (int q = 0; q < 4; ++q) {
      const int e   = tid + (q << 8);
      const int b_  = e >> 4, j = e & 15;
      const int hid = (s << 4) + j;
      const size_t xb = (size_t)b_ * G4H + hid;
      float gi = s_g[b_ * 64 +      j] + (float)xgr[xb];
      float gf = s_g[b_ * 64 + 16 + j] + (float)xgr[xb + 512];
      float gg = s_g[b_ * 64 + 32 + j] + (float)xgr[xb + 1024];
      float go = s_g[b_ * 64 + 48 + j] + (float)xgr[xb + 1536];
      gi = sigm_(gi); gf = sigm_(gf); gg = tanh_(gg); go = sigm_(go);
      const float cn = gf * c_reg[q] + gi * gg;
      c_reg[q] = cn;
      const float hv = go * tanh_(cn);
      hn[(size_t)b_ * HID + hid] = (_Float16)hv;
      if (t == T_STEPS - 1) {
        hT[(size_t)b_ * HID + hid] = hv;
        cT[(size_t)b_ * HID + hid] = cn;
      }
    }
    grid_barrier(bar);   // h_{t+1} complete device-wide before next step
  }
}

// ---------------------------------------------------------------------------
// Launch
// ---------------------------------------------------------------------------
extern "C" void kernel_launch(void* const* d_in, const int* in_sizes, int n_in,
                              void* d_out, int out_size, void* d_ws, size_t ws_size,
                              hipStream_t stream) {
  const float* inputs = (const float*)d_in[0];
  const float* W_ih   = (const float*)d_in[1];
  const float* W_hh   = (const float*)d_in[2];
  const float* b_ih   = (const float*)d_in[3];
  const float* b_hh   = (const float*)d_in[4];
  const float* W_out  = (const float*)d_in[5];
  const float* b_out  = (const float*)d_in[6];
  float* out = (float*)d_out;

  char* ws = (char*)d_ws;
  size_t o = 0;
  unsigned* bar = (unsigned*)(ws + o);              o += 256;
  _Float16* xf    = (_Float16*)(ws + o);            o += (size_t)T_STEPS * BATCH * IN_DIM * 2;
  _Float16* wihP  = (_Float16*)(ws + o);            o += (size_t)G4H * IN_DIM * 2;
  _Float16* whhP  = (_Float16*)(ws + o);            o += (size_t)G4H * HID * 2;
  _Float16* woutP = (_Float16*)(ws + o);            o += (size_t)OUT_DIM * HID * 2;
  _Float16* xgw   = (_Float16*)(ws + o);            o += (size_t)T_STEPS * BATCH * G4H * 2;
  _Float16* hseq  = (_Float16*)(ws + o);            o += (size_t)(T_STEPS + 1) * BATCH * HID * 2;

  init_state<<<128, 256, 0, stream>>>(bar, hseq);

  const int n4 = T_STEPS * BATCH * IN_DIM / 4;
  cvt_f32_to_f16<<<(n4 + 255) / 256, 256, 0, stream>>>(inputs, xf, n4);

  pack_wB<<<(G4H * IN_DIM + 255) / 256, 256, 0, stream>>>(W_ih, wihP, G4H, IN_DIM);
  pack_wB<<<(G4H * HID + 255) / 256, 256, 0, stream>>>(W_hh, whhP, G4H, HID);
  pack_wB<<<(OUT_DIM * HID + 255) / 256, 256, 0, stream>>>(W_out, woutP, OUT_DIM, HID);

  // x_gates = X @ W_ih^T + b_ih + b_hh  -> f16 [T*B, 4H]
  dim3 g1(T_STEPS * BATCH / 64, G4H / 128);
  wmma_gemm<<<g1, 128, 0, stream>>>(xf, wihP, nullptr, xgw, b_ih, b_hh,
                                    T_STEPS * BATCH, G4H, IN_DIM);

  float* hT = out + (size_t)T_STEPS * BATCH * OUT_DIM;
  float* cT = hT + (size_t)BATCH * HID;
  lstm_recurrent<<<NWG_REC, 256, 0, stream>>>(xgw, whhP, hseq, hT, cT, bar);

  // outputs = hs @ W_out^T + b_out  (hs = hseq[1..T])
  dim3 g3(T_STEPS * BATCH / 64, OUT_DIM / 128);
  wmma_gemm<<<g3, 128, 0, stream>>>(hseq + (size_t)BATCH * HID, woutP, out, nullptr,
                                    b_out, nullptr, T_STEPS * BATCH, OUT_DIM, HID);
}
